// LundNet_Ntrk_Plus_12996571038299
// MI455X (gfx1250) — matrix operand order repaired and checked by reading, not compiled, not run on hardware
//
#include <hip/hip_runtime.h>
#include <hip/hip_bf16.h>
#include <math.h>

typedef __attribute__((ext_vector_type(16))) _Float16 v16h;
typedef __attribute__((ext_vector_type(8)))  _Float16 v8h;
typedef __attribute__((ext_vector_type(8)))  float    v8f;

#define THREADS 256

// ---------------------------------------------------------------------------
// Utility kernels
// ---------------------------------------------------------------------------
__global__ void zero_f32_kernel(float* __restrict__ p, size_t n) {
    size_t i = (size_t)blockIdx.x * blockDim.x + threadIdx.x;
    if (i < n) p[i] = 0.0f;
}

__global__ void f32_to_f16_kernel(const float* __restrict__ in,
                                  _Float16* __restrict__ out, size_t n) {
    size_t i = (size_t)blockIdx.x * blockDim.x + threadIdx.x;
    if (i < n) out[i] = (_Float16)in[i];
}

// ---------------------------------------------------------------------------
// Fold BN scale into weights AND pre-pack into the WMMA B-fragment layout.
// Source W row-major [din, dout] f32; dest Wp holds Kpad*dout f16 arranged as
//   Wp[ ((kt*nTiles + nt)*32 + lane)*16 + e ]
// where kt = k/32, nt = j/16, lane = (j&15) + 16*khalf, khalf = K-half,
// e = fragment element:  k = kt*32 + 8*khalf + (e&7) + 16*(e>>3).
// This lets each GEMM lane fetch its whole B fragment as one contiguous v16h.
// ---------------------------------------------------------------------------
__global__ void fold_w_pack_kernel(const float* __restrict__ W,
                                   const float* __restrict__ g,
                                   const float* __restrict__ v,
                                   int din, int Kpad, int dout,
                                   _Float16* __restrict__ Wp) {
    int i = blockIdx.x * blockDim.x + threadIdx.x;          // packed index
    if (i >= Kpad * dout) return;
    int e     = i & 15;
    int lane  = (i >> 4) & 31;
    int tile  = i >> 9;                                     // 512 halves/tile
    int nTiles = dout >> 4;
    int kt    = tile / nTiles;
    int nt    = tile - kt * nTiles;
    int khalf = lane >> 4;
    int j     = nt * 16 + (lane & 15);                      // output column
    int k     = kt * 32 + 8 * khalf + (e & 7) + 16 * (e >> 3); // K row
    float s   = g[j] * rsqrtf(v[j] + 1e-5f);
    Wp[i] = (_Float16)((k < din) ? W[(size_t)k * dout + j] * s : 0.0f);
}

__global__ void fold_b_kernel(const float* __restrict__ b,
                              const float* __restrict__ g,
                              const float* __restrict__ beta,
                              const float* __restrict__ m,
                              const float* __restrict__ v,
                              int dout, float* __restrict__ bf) {
    int j = blockIdx.x * blockDim.x + threadIdx.x;
    if (j >= dout) return;
    float s = g[j] * rsqrtf(v[j] + 1e-5f);
    bf[j] = (b[j] - m[j]) * s + beta[j];
}

// ---------------------------------------------------------------------------
// EdgeConv message gather:  msg[e, 0:d] = x[dst[e]], msg[e, d:2d] = x[src]-x[dst]
// zero-padded to Kpad columns, output f16 row-major.
// ---------------------------------------------------------------------------
__global__ void gather_msg_kernel(const float* __restrict__ xfeat, int ld, int colIn,
                                  const int* __restrict__ src,
                                  const int* __restrict__ dst,
                                  int E, int d, int Kpad,
                                  _Float16* __restrict__ msg) {
    size_t t = (size_t)blockIdx.x * blockDim.x + threadIdx.x;
    size_t total = (size_t)E * Kpad;
    if (t >= total) return;
    int e = (int)(t / Kpad);
    int c = (int)(t % Kpad);
    float val = 0.0f;
    if (c < d) {
        val = xfeat[(size_t)dst[e] * ld + colIn + c];
    } else if (c < 2 * d) {
        int cc = c - d;
        float xi = xfeat[(size_t)dst[e] * ld + colIn + cc];
        float xj = xfeat[(size_t)src[e] * ld + colIn + cc];
        val = xj - xi;
    }
    msg[t] = (_Float16)val;
}

// ---------------------------------------------------------------------------
// WMMA GEMM:  C[M,N] = relu(A[M,K] @ B[K,N] + bias)
//   A: f16 row-major (lda = K); B: f16 PRE-PACKED fragment layout (see above)
//   OUT_MODE 0: write f16 row-major (ld = N)
//   OUT_MODE 1: atomic-add f32 into outF[dstIdx[m]*outLd + colOff + n]
//   OUT_MODE 2: write f32 into outF[m*outLd + colOff + n]
// One wave computes one 16x16 tile via v_wmma_f32_16x16x32_f16.
// M, N multiples of 16; K multiple of 32. Guard is wave-uniform (EXEC all-1s).
// Inner loop: two v8h A loads + one v16h B load + 1 WMMA.
// ---------------------------------------------------------------------------
template <int OUT_MODE>
__global__ __launch_bounds__(THREADS)
void wmma_gemm_kernel(const _Float16* __restrict__ A,
                      const _Float16* __restrict__ Bp,
                      const float* __restrict__ bias,
                      int M, int K, int N,
                      _Float16* __restrict__ outH,
                      float* __restrict__ outF,
                      const int* __restrict__ dstIdx,
                      int outLd, int colOff) {
    int wave  = (blockIdx.x * blockDim.x + threadIdx.x) >> 5;
    int lane  = threadIdx.x & 31;
    int nT    = N >> 4;
    int mTile = wave / nT;
    int nTile = wave - mTile * nT;
    if (mTile * 16 >= M) return;          // wave-uniform

    int l15   = lane & 15;
    int khalf = lane >> 4;
    int row   = mTile * 16 + l15;         // A row for this lane
    int col   = nTile * 16 + l15;         // C column for this lane

    const _Float16* aRow  = A + (size_t)row * K + 8 * khalf;
    const _Float16* bBase = Bp + ((size_t)nTile * 32 + lane) * 16;
    size_t bTileStride = (size_t)nT * 512;        // halves per K-tile of Bp

    v8f acc = {};
    for (int k0 = 0; k0 < K; k0 += 32) {
        // --- A fragment (16x32 f16): two contiguous 16B loads per lane
        const _Float16* ap = aRow + k0;
        v8h a_lo = *(const v8h*)(ap);        // K = k0+8*khalf .. +7
        v8h a_hi = *(const v8h*)(ap + 16);   // K = k0+16+8*khalf .. +7
        if (k0 + 32 < K) __builtin_prefetch(ap + 32, 0, 1);  // next A slice
        v16h a;
        #pragma unroll
        for (int i = 0; i < 8; i++) { a[i] = a_lo[i]; a[8 + i] = a_hi[i]; }

        // --- B fragment: one contiguous 32B load from pre-packed weights
        v16h b = *(const v16h*)(bBase + (size_t)(k0 >> 5) * bTileStride);

        acc = __builtin_amdgcn_wmma_f32_16x16x32_f16(
            /*neg_a=*/false, a, /*neg_b=*/false, b,
            /*c_mod=*/(short)0, acc, /*reuse_a=*/false, /*reuse_b=*/false);
    }

    // Epilogue: bias + ReLU; C layout: VGPR r -> M = r + 8*khalf, N = lane&15
    float bs = bias[col];
    #pragma unroll
    for (int r = 0; r < 8; r++) {
        int m  = mTile * 16 + r + 8 * khalf;
        float v = acc[r] + bs;
        v = v > 0.0f ? v : 0.0f;
        if (OUT_MODE == 0) {
            outH[(size_t)m * N + col] = (_Float16)v;
        } else if (OUT_MODE == 1) {
            atomicAdd(&outF[(size_t)dstIdx[m] * outLd + colOff + col], v);
        } else {
            outF[(size_t)m * outLd + colOff + col] = v;
        }
    }
}

// ---------------------------------------------------------------------------
// Pooling + heads
// ---------------------------------------------------------------------------
__global__ void pool_sum_kernel(const float* __restrict__ h,
                                const int* __restrict__ batch,
                                float* __restrict__ pooled, int N) {
    size_t t = (size_t)blockIdx.x * blockDim.x + threadIdx.x;
    if (t >= (size_t)N * 384) return;
    int i = (int)(t / 384), c = (int)(t % 384);
    atomicAdd(&pooled[(size_t)batch[i] * 384 + c], h[t]);
}

__global__ void count_kernel(const int* __restrict__ batch,
                             float* __restrict__ counts, int N) {
    int i = blockIdx.x * blockDim.x + threadIdx.x;
    if (i < N) atomicAdd(&counts[batch[i]], 1.0f);
}

__global__ void build_g_kernel(const float* __restrict__ pooled,
                               const float* __restrict__ counts,
                               const float* __restrict__ gf,
                               float* __restrict__ g, int B) {
    int t = blockIdx.x * blockDim.x + threadIdx.x;
    if (t >= B * 394) return;
    int b = t / 394, c = t % 394;
    g[t] = (c < 384) ? pooled[(size_t)b * 384 + c] / fmaxf(counts[b], 1.0f)
                     : gf[(size_t)b * 10 + (c - 384)];
}

// act: 1 = relu, 2 = sigmoid
__global__ void dense_kernel(const float* __restrict__ in,
                             const float* __restrict__ W,
                             const float* __restrict__ bias,
                             float* __restrict__ out,
                             int M, int K, int N, int act) {
    int t = blockIdx.x * blockDim.x + threadIdx.x;
    if (t >= M * N) return;
    int row = t / N, j = t % N;
    float acc = bias[j];
    const float* ip = in + (size_t)row * K;
    for (int k = 0; k < K; k++) acc += ip[k] * W[(size_t)k * N + j];
    if (act == 1) acc = fmaxf(acc, 0.0f);
    else if (act == 2) acc = 1.0f / (1.0f + __expf(-acc));
    out[t] = acc;
}

// ---------------------------------------------------------------------------
// Host launch
// ---------------------------------------------------------------------------
static inline unsigned nblk(size_t n) { return (unsigned)((n + THREADS - 1) / THREADS); }

extern "C" void kernel_launch(void* const* d_in, const int* in_sizes, int n_in,
                              void* d_out, int out_size, void* d_ws, size_t ws_size,
                              hipStream_t stream) {
    const float* x      = (const float*)d_in[0];
    const int*   ei     = (const int*)d_in[1];
    const int*   batch  = (const int*)d_in[2];
    const float* gfeat  = (const float*)d_in[3];
    const int N = in_sizes[0] / 3;
    const int E = in_sizes[1] / 2;
    const int B = in_sizes[3] / 10;
    const int* src = ei;
    const int* dst = ei + E;

    // params flattened alphabetically: convs(6 x [W1,W2,b1,b2,beta1,beta2,g1,g2,m1,m2,v1,v2]),
    // lin[W,b], seq1[W,b,beta,g,m,v], seq2[W,b], seq3[W,b]
    const float* cp[6][12];
    for (int c = 0; c < 6; c++)
        for (int j = 0; j < 12; j++) cp[c][j] = (const float*)d_in[4 + 12 * c + j];
    const float* linW   = (const float*)d_in[76];
    const float* linB   = (const float*)d_in[77];
    const float* s1W    = (const float*)d_in[78];
    const float* s1b    = (const float*)d_in[79];
    const float* s1beta = (const float*)d_in[80];
    const float* s1g    = (const float*)d_in[81];
    const float* s1m    = (const float*)d_in[82];
    const float* s1v    = (const float*)d_in[83];
    const float* s2W    = (const float*)d_in[84];
    const float* s2b    = (const float*)d_in[85];
    const float* s3W    = (const float*)d_in[86];
    const float* s3b    = (const float*)d_in[87];

    // --- workspace bump allocator
    char* wsb = (char*)d_ws;
    size_t off = 0;
    auto alloc = [&](size_t bytes) -> void* {
        void* p = wsb + off;
        off = (off + bytes + 255) & ~(size_t)255;
        return p;
    };
    float*    Ycat   = (float*)alloc((size_t)N * 448 * 4);
    _Float16* msg    = (_Float16*)alloc((size_t)E * 256 * 2);
    _Float16* h1     = (_Float16*)alloc((size_t)E * 128 * 2);
    float*    h384   = (float*)alloc((size_t)N * 384 * 4);
    float*    pooled = (float*)alloc((size_t)B * 384 * 4);
    float*    counts = (float*)alloc((size_t)B * 4);
    float*    gbuf   = (float*)alloc((size_t)B * 394 * 4);
    float*    g2     = (float*)alloc((size_t)B * 256 * 4);
    float*    g3     = (float*)alloc((size_t)B * 30 * 4);
    _Float16* catF16 = msg;  // reuse: N*448*2 <= E*256*2

    // layer dims: node width d, padded K, hidden d1, out d2, concat offsets
    const int dN[6]     = {3, 32, 32, 64, 64, 128};
    const int Kpad[6]   = {32, 64, 64, 128, 128, 256};
    const int d1[6]     = {32, 32, 64, 64, 128, 128};
    const int d2[6]     = {32, 32, 64, 64, 128, 128};
    const int colIn[6]  = {0, 0, 32, 64, 128, 192};   // input slice in Ycat (l>=1)
    const int colOut[6] = {0, 32, 64, 128, 192, 320};

    _Float16* Wf1[6]; float* bf1[6]; _Float16* Wf2[6]; float* bf2[6];
    for (int l = 0; l < 6; l++) {
        Wf1[l] = (_Float16*)alloc((size_t)Kpad[l] * d1[l] * 2);
        bf1[l] = (float*)alloc((size_t)d1[l] * 4);
        Wf2[l] = (_Float16*)alloc((size_t)d1[l] * d2[l] * 2);
        bf2[l] = (float*)alloc((size_t)d2[l] * 4);
    }
    _Float16* Ws1 = (_Float16*)alloc((size_t)448 * 384 * 2);
    float*    bs1 = (float*)alloc((size_t)384 * 4);
    (void)ws_size; (void)n_in; (void)out_size;

    // --- zero accumulators
    zero_f32_kernel<<<nblk((size_t)N * 448), THREADS, 0, stream>>>(Ycat, (size_t)N * 448);
    zero_f32_kernel<<<nblk((size_t)B * 384), THREADS, 0, stream>>>(pooled, (size_t)B * 384);
    zero_f32_kernel<<<nblk((size_t)B), THREADS, 0, stream>>>(counts, (size_t)B);

    // --- fold BN into weights (f16, pre-packed fragments) and biases (f32)
    for (int l = 0; l < 6; l++) {
        const float *W1 = cp[l][0], *W2 = cp[l][1], *b1 = cp[l][2], *b2 = cp[l][3];
        const float *be1 = cp[l][4], *be2 = cp[l][5], *g1 = cp[l][6], *g2p = cp[l][7];
        const float *m1 = cp[l][8], *m2 = cp[l][9], *v1 = cp[l][10], *v2 = cp[l][11];
        int din = 2 * dN[l];
        fold_w_pack_kernel<<<nblk((size_t)Kpad[l] * d1[l]), THREADS, 0, stream>>>(
            W1, g1, v1, din, Kpad[l], d1[l], Wf1[l]);
        fold_b_kernel<<<nblk((size_t)d1[l]), THREADS, 0, stream>>>(
            b1, g1, be1, m1, v1, d1[l], bf1[l]);
        fold_w_pack_kernel<<<nblk((size_t)d1[l] * d2[l]), THREADS, 0, stream>>>(
            W2, g2p, v2, d1[l], d1[l], d2[l], Wf2[l]);
        fold_b_kernel<<<nblk((size_t)d2[l]), THREADS, 0, stream>>>(
            b2, g2p, be2, m2, v2, d2[l], bf2[l]);
    }
    fold_w_pack_kernel<<<nblk((size_t)448 * 384), THREADS, 0, stream>>>(
        s1W, s1g, s1v, 448, 448, 384, Ws1);
    fold_b_kernel<<<nblk((size_t)384), THREADS, 0, stream>>>(
        s1b, s1g, s1beta, s1m, s1v, 384, bs1);

    // --- EdgeConv stack
    for (int l = 0; l < 6; l++) {
        const float* xin = (l == 0) ? x : Ycat;
        int ld = (l == 0) ? 3 : 448;
        gather_msg_kernel<<<nblk((size_t)E * Kpad[l]), THREADS, 0, stream>>>(
            xin, ld, colIn[l], src, dst, E, dN[l], Kpad[l], msg);

        // GEMM1: h1 = relu(bn1(msg @ W1 + b1))  -> f16 [E, d1]
        {
            long waves = (long)(E / 16) * (d1[l] / 16);
            wmma_gemm_kernel<0><<<(unsigned)((waves + 7) / 8), THREADS, 0, stream>>>(
                msg, Wf1[l], bf1[l], E, Kpad[l], d1[l], h1, nullptr, nullptr, 0, 0);
        }
        // GEMM2: relu(bn2(h1 @ W2 + b2)) scatter-added into Ycat slice
        {
            long waves = (long)(E / 16) * (d2[l] / 16);
            wmma_gemm_kernel<1><<<(unsigned)((waves + 7) / 8), THREADS, 0, stream>>>(
                h1, Wf2[l], bf2[l], E, d1[l], d2[l], nullptr, Ycat, dst, 448, colOut[l]);
        }
    }

    // --- seq1: h384 = relu(bn(Ycat @ W + b)), [N,448] @ [448,384]
    f32_to_f16_kernel<<<nblk((size_t)N * 448), THREADS, 0, stream>>>(
        Ycat, catF16, (size_t)N * 448);
    {
        long waves = (long)(N / 16) * (384 / 16);
        wmma_gemm_kernel<2><<<(unsigned)((waves + 7) / 8), THREADS, 0, stream>>>(
            catF16, Ws1, bs1, N, 448, 384, nullptr, h384, nullptr, 384, 0);
    }

    // --- global mean pool + graph feature concat
    pool_sum_kernel<<<nblk((size_t)N * 384), THREADS, 0, stream>>>(h384, batch, pooled, N);
    count_kernel<<<nblk((size_t)N), THREADS, 0, stream>>>(batch, counts, N);
    build_g_kernel<<<nblk((size_t)B * 394), THREADS, 0, stream>>>(pooled, counts, gfeat, gbuf, B);

    // --- heads
    dense_kernel<<<nblk((size_t)B * 256), THREADS, 0, stream>>>(gbuf, s2W, s2b, g2, B, 394, 256, 1);
    dense_kernel<<<nblk((size_t)B * 30),  THREADS, 0, stream>>>(g2, s3W, s3b, g3, B, 256, 30, 1);
    dense_kernel<<<nblk((size_t)B),       THREADS, 0, stream>>>(g3, linW, linB, (float*)d_out, B, 30, 1, 2);
}
